// GPT_74706661146762
// MI455X (gfx1250) — compile-verified, matrix operand order checked by reading, not run on hardware
//
#include <hip/hip_runtime.h>
#include <math.h>

// ---------------- model dims ----------------
#define T_SEQ   1024
#define DMODEL  768
#define HHEADS  8
#define HDIM    96
#define DFF     3072
#define NVOCAB  32000
#define NBLK    10          // M
#define NROUTE  11          // M+1
#define RFEAT   770         // D+2

typedef __bf16 bf16_t;
typedef __attribute__((ext_vector_type(16))) __bf16 v16bf;
typedef __attribute__((ext_vector_type(8)))  __bf16 v8bf;
typedef __attribute__((ext_vector_type(8)))  float  v8f;
typedef __attribute__((ext_vector_type(4)))  float  v4f;

struct RState { int cur; int done; float cnt; };

// ---------------- WMMA helpers ----------------
__device__ __forceinline__ v8f wmma_bf16(v16bf a, v16bf b, v8f c) {
  // D = A(16x32) * B(32x16) + C(16x16 f32)
  return __builtin_amdgcn_wmma_f32_16x16x32_bf16(false, a, false, b, (short)0, c,
                                                 false, false);
}

// A-fragment (16x32, row-major source): lane holds row m=lane&15,
// elements e<8 -> k = 8*half + e ; e>=8 -> k = 16 + 8*half + (e-8).
// B-fragment for C = X * W^T is identical with "row" = output column n.
__device__ __forceinline__ v16bf load_frag_bf16(const bf16_t* __restrict__ base,
                                                int ld, int row, int k0, int lane) {
  int half = (lane >> 4) & 1;
  const bf16_t* p = base + (size_t)row * ld + k0 + 8 * half;
  v8bf lo = *(const v8bf*)p;
  v8bf hi = *(const v8bf*)(p + 16);
  v16bf a;
#pragma unroll
  for (int i = 0; i < 8; ++i) { a[i] = lo[i]; a[i + 8] = hi[i]; }
  return a;
}

// ---------------- generic WMMA GEMM: C[1024 x N] = act( A_bf16[1024 x K] @ W_f32[N x K]^T )
// Block tile 256(M) x 64(N); W tile staged once per workgroup in LDS as bf16.
// Each wave owns 32 rows x 64 cols: 2 A-frags x 4 B-frags = 8 WMMAs per k-step.
// mode 0: f32 out ; 1: f32 out += Cin (residual) ; 2: relu^2 -> bf16 out ; 3: 30*tanh(x/30) f32
#define WPAD 40   // padded row length (bf16) to dodge bank conflicts; 80B rows
__global__ __launch_bounds__(256)
void gemm_wmma_kernel(const bf16_t* __restrict__ A, const float* __restrict__ Wbase,
                      long long wstride, const int* __restrict__ curPtr,
                      int N, int K, const float* __restrict__ Cin,
                      void* __restrict__ Cout, int mode)
{
  __shared__ bf16_t sW[64][WPAD];             // 64 output cols x 32 k (bf16)
  const int tid  = threadIdx.x;
  const int lane = tid & 31;
  const int wave = tid >> 5;                  // 8 waves: 256 rows / 32
  const float* W = Wbase + (curPtr ? (size_t)curPtr[0] * (size_t)wstride : 0);
  const int tileN = blockIdx.x * 64;
  const int tileM = blockIdx.y * 256;
  const int n     = lane & 15;
  const int half  = (lane >> 4) & 1;
  const int rowA0 = tileM + wave * 32 + n;
  const int rowA1 = rowA0 + 16;

  // cooperative staging map: thread -> (row in N-tile, 8-wide k chunk)
  const int wr = tid >> 2;                    // 0..63
  const int wk = (tid & 3) * 8;               // 0,8,16,24

  v8f acc[4][2] = {};
  for (int ks = 0; ks < K; ks += 32) {
    const float* wp = W + (size_t)(tileN + wr) * K + ks + wk;
    v4f f0 = *(const v4f*)wp;
    v4f f1 = *(const v4f*)(wp + 4);
    v8bf wv;
#pragma unroll
    for (int i = 0; i < 4; ++i) { wv[i] = (bf16_t)f0[i]; wv[4 + i] = (bf16_t)f1[i]; }
    *(v8bf*)&sW[wr][wk] = wv;
    if (ks + 32 < K) __builtin_prefetch(wp + 32, 0, 1);   // global_prefetch_b8
    __syncthreads();

    v16bf a0 = load_frag_bf16(A, K, rowA0, ks, lane);
    v16bf a1 = load_frag_bf16(A, K, rowA1, ks, lane);
#pragma unroll
    for (int j = 0; j < 4; ++j) {
      const bf16_t* p = &sW[j * 16 + n][8 * half];
      v8bf lo = *(const v8bf*)p;
      v8bf hi = *(const v8bf*)(p + 16);
      v16bf b;
#pragma unroll
      for (int i = 0; i < 8; ++i) { b[i] = lo[i]; b[i + 8] = hi[i]; }
      acc[j][0] = wmma_bf16(a0, b, acc[j][0]);
      acc[j][1] = wmma_bf16(a1, b, acc[j][1]);
    }
    __syncthreads();
  }
#pragma unroll
  for (int j = 0; j < 4; ++j) {
    int col = tileN + j * 16 + n;
#pragma unroll
    for (int m2 = 0; m2 < 2; ++m2) {
#pragma unroll
      for (int r = 0; r < 8; ++r) {
        int row = tileM + wave * 32 + m2 * 16 + half * 8 + r;
        size_t idx = (size_t)row * N + col;
        float v = acc[j][m2][r];
        if (mode == 0)      ((float*)Cout)[idx] = v;
        else if (mode == 1) ((float*)Cout)[idx] = Cin[idx] + v;
        else if (mode == 2) { float t = v > 0.f ? v : 0.f; ((bf16_t*)Cout)[idx] = (bf16_t)(t * t); }
        else                ((float*)Cout)[idx] = 30.f * tanhf(v * (1.f / 30.f));
      }
    }
  }
}

// ---------------- rope table ----------------
__global__ void rope_table_kernel(float* __restrict__ cosb, float* __restrict__ sinb)
{
  int id = blockIdx.x * 64 + threadIdx.x;
  if (id >= T_SEQ * (HDIM / 2)) return;
  int t = id / (HDIM / 2), i = id % (HDIM / 2);
  float inv = (i < HDIM / 4) ? powf(1.0f / 1024.0f, (float)i / (float)(HDIM / 4 - 1)) : 0.f;
  float th = (float)t * inv;
  cosb[id] = cosf(th);
  sinb[id] = sinf(th);
}

// ---------------- embed gather + rms ----------------
__global__ __launch_bounds__(256)
void embed_kernel(const int* __restrict__ tok, const float* __restrict__ Ew,
                  const float* __restrict__ Vw, float* __restrict__ x,
                  float* __restrict__ x0, float* __restrict__ ve)
{
  __shared__ float red[8];
  int t = blockIdx.x;
  int tk = tok[t];
  const float* e = Ew + (size_t)tk * DMODEL;
  float ss = 0.f;
  for (int i = threadIdx.x; i < DMODEL; i += 256) { float v = e[i]; ss += v * v; }
  for (int off = 16; off; off >>= 1) ss += __shfl_xor(ss, off, 32);
  if ((threadIdx.x & 31) == 0) red[threadIdx.x >> 5] = ss;
  __syncthreads();
  float tot = 0.f;
#pragma unroll
  for (int i = 0; i < 8; ++i) tot += red[i];
  float s = rsqrtf(tot / (float)DMODEL + 1e-6f);
  const float* vr = Vw + (size_t)tk * DMODEL;
  for (int i = threadIdx.x; i < DMODEL; i += 256) {
    float v = e[i] * s;
    size_t idx = (size_t)t * DMODEL + i;
    x[idx] = v; x0[idx] = v; ve[idx] = vr[i];
  }
}

// ---------------- row RMS -> bf16 ----------------
__global__ __launch_bounds__(256)
void rms_row_kernel(const float* __restrict__ X, bf16_t* __restrict__ Y, int D)
{
  __shared__ float red[8];
  int row = blockIdx.x;
  const float* x = X + (size_t)row * D;
  float ss = 0.f;
  for (int i = threadIdx.x; i < D; i += 256) { float v = x[i]; ss += v * v; }
  for (int off = 16; off; off >>= 1) ss += __shfl_xor(ss, off, 32);
  if ((threadIdx.x & 31) == 0) red[threadIdx.x >> 5] = ss;
  __syncthreads();
  float tot = 0.f;
#pragma unroll
  for (int i = 0; i < 8; ++i) tot += red[i];
  float s = rsqrtf(tot / (float)D + 1e-6f);
  for (int i = threadIdx.x; i < D; i += 256)
    Y[(size_t)row * D + i] = (bf16_t)(x[i] * s);
}

// ---------------- per-head RMS + RoPE -> bf16 [H][T][HD] ----------------
__global__ __launch_bounds__(32)
void rope_head_kernel(const float* __restrict__ X, const float* __restrict__ cosb,
                      const float* __restrict__ sinb, bf16_t* __restrict__ out)
{
  int id = blockIdx.x;
  int t = id >> 3, h = id & 7;
  const float* x = X + (size_t)t * DMODEL + h * HDIM;
  int lane = threadIdx.x;
  float ss = 0.f;
#pragma unroll
  for (int i = lane; i < HDIM; i += 32) { float v = x[i]; ss += v * v; }
  for (int off = 16; off; off >>= 1) ss += __shfl_xor(ss, off, 32);
  float s = rsqrtf(ss / (float)HDIM + 1e-6f);
  bf16_t* o = out + ((size_t)h * T_SEQ + t) * HDIM;
  for (int p = lane; p < HDIM / 2; p += 32) {
    float x1 = x[p] * s, x2 = x[p + HDIM / 2] * s;
    float c = cosb[t * (HDIM / 2) + p], sn = sinb[t * (HDIM / 2) + p];
    o[p]            = (bf16_t)( x1 * c + x2 * sn);
    o[p + HDIM / 2] = (bf16_t)(-x1 * sn + x2 * c);
  }
}

// ---------------- v = lam0*v (+ lam1*ve) -> bf16 [H][T][HD] ----------------
__global__ void vprep_kernel(const float* __restrict__ V, const float* __restrict__ VE,
                             const float* __restrict__ lamBase, const int* __restrict__ curPtr,
                             bf16_t* __restrict__ out)
{
  int id = blockIdx.x * 256 + threadIdx.x;
  if (id >= T_SEQ * DMODEL) return;
  const float* lam = lamBase + (curPtr ? curPtr[0] * 2 : 0);
  int t = id / DMODEL, d = id % DMODEL, h = d / HDIM, hd = d % HDIM;
  float v = lam[0] * V[id];
  if (VE) v += lam[1] * VE[id];
  out[((size_t)h * T_SEQ + t) * HDIM + hd] = (bf16_t)v;
}

// ---------------- flash attention (windowed causal), WMMA QK^T and P*V ----------------
__global__ __launch_bounds__(32)
void attn_kernel(const bf16_t* __restrict__ qh, const bf16_t* __restrict__ kh,
                 const bf16_t* __restrict__ vh, bf16_t* __restrict__ out,
                 const int* __restrict__ wbPtr)
{
  __shared__ float  sP[16][32];       // probabilities (C-layout -> A-layout staging)
  __shared__ bf16_t sV[32][HDIM];     // 32 keys x 96, staged per key tile
  const int lane = threadIdx.x & 31;
  const int qt = blockIdx.x;          // 16-query tile
  const int h  = blockIdx.y;          // head
  const int half = (lane >> 4) & 1, n = lane & 15;
  const bf16_t* Q  = qh + (size_t)h * T_SEQ * HDIM;
  const bf16_t* Km = kh + (size_t)h * T_SEQ * HDIM;
  const bf16_t* Vm = vh + (size_t)h * T_SEQ * HDIM;
  const int q0 = qt * 16;
  const int wb = wbPtr[0];
  int qb = q0 >> 7;
  int kb0 = qb - wb + 1; if (kb0 < 0) kb0 = 0;
  const int kstart = kb0 * 128;
  const int kend   = q0 + 16;         // exclusive; masking covers the K-tile overhang
  const float scale = 0.1020620726159658f; // 1/sqrt(96)

  v16bf qf[3];
#pragma unroll
  for (int c = 0; c < 3; ++c) qf[c] = load_frag_bf16(Q, HDIM, q0 + n, c * 32, lane);

  v8f o[6] = {};
  float mrow[8], lrow[8];
#pragma unroll
  for (int r = 0; r < 8; ++r) { mrow[r] = -3.0e38f; lrow[r] = 0.f; }

  for (int ks = kstart; ks < kend; ks += 32) {
    // ---- scores S = Q K^T for 16 x 32 keys ----
    v8f s0 = {}, s1 = {};
#pragma unroll
    for (int c = 0; c < 3; ++c) {
      v16bf b0 = load_frag_bf16(Km, HDIM, ks + n, c * 32, lane);
      v16bf b1 = load_frag_bf16(Km, HDIM, ks + 16 + n, c * 32, lane);
      s0 = wmma_bf16(qf[c], b0, s0);
      s1 = wmma_bf16(qf[c], b1, s1);
    }
    // ---- mask + online softmax ----
#pragma unroll
    for (int r = 0; r < 8; ++r) {
      int qrow = q0 + half * 8 + r;
      int key0 = ks + n, key1 = ks + 16 + n;
      float v0 = s0[r] * scale, v1 = s1[r] * scale;
      bool ok0 = (key0 <= qrow) && ((qrow >> 7) - (key0 >> 7) < wb);
      bool ok1 = (key1 <= qrow) && ((qrow >> 7) - (key1 >> 7) < wb);
      v0 = ok0 ? v0 : -1e30f;
      v1 = ok1 ? v1 : -1e30f;
      float mx = fmaxf(v0, v1);
#pragma unroll
      for (int off = 1; off < 16; off <<= 1) mx = fmaxf(mx, __shfl_xor(mx, off, 32));
      float mnew  = fmaxf(mrow[r], mx);
      float alpha = __expf(mrow[r] - mnew);
      float p0 = __expf(v0 - mnew), p1 = __expf(v1 - mnew);
      float ps = p0 + p1;
#pragma unroll
      for (int off = 1; off < 16; off <<= 1) ps += __shfl_xor(ps, off, 32);
      lrow[r] = lrow[r] * alpha + ps;
      mrow[r] = mnew;
#pragma unroll
      for (int j = 0; j < 6; ++j) o[j][r] *= alpha;   // same row layout as scores
      sP[half * 8 + r][n]      = p0;
      sP[half * 8 + r][16 + n] = p1;
    }
    // ---- stage V tile (32 keys x 96) to LDS: lane copies one key row ----
    {
      const bf16_t* vr = Vm + (size_t)(ks + lane) * HDIM;
#pragma unroll
      for (int c = 0; c < HDIM / 8; ++c)
        *(v8bf*)&sV[lane][c * 8] = *(const v8bf*)(vr + c * 8);
    }
    __syncthreads();
    // ---- P fragment (A-layout) from LDS ----
    v16bf pf;
#pragma unroll
    for (int i = 0; i < 8; ++i) {
      pf[i]     = (bf16_t)sP[n][8 * half + i];
      pf[i + 8] = (bf16_t)sP[n][16 + 8 * half + i];
    }
    // ---- O += P @ V  (B frags gathered from LDS) ----
#pragma unroll
    for (int j = 0; j < 6; ++j) {
      int col = j * 16 + n;
      v16bf vf;
#pragma unroll
      for (int i = 0; i < 8; ++i) {
        vf[i]     = sV[8 * half + i][col];
        vf[i + 8] = sV[16 + 8 * half + i][col];
      }
      o[j] = wmma_bf16(pf, vf, o[j]);
    }
    __syncthreads();
  }
#pragma unroll
  for (int r = 0; r < 8; ++r) {
    float inv = 1.f / lrow[r];
    int row = q0 + half * 8 + r;
#pragma unroll
    for (int j = 0; j < 6; ++j)
      out[(size_t)row * DMODEL + h * HDIM + j * 16 + n] = (bf16_t)(o[j][r] * inv);
  }
}

// ---------------- h = lam0*x + lam1*x0 ----------------
__global__ void mix_kernel(float* __restrict__ x, const float* __restrict__ x0,
                           const float* __restrict__ lam)
{
  int id = blockIdx.x * 256 + threadIdx.x;
  if (id >= T_SEQ * DMODEL) return;
  x[id] = lam[0] * x[id] + lam[1] * x0[id];
}

// ---------------- router state ----------------
__global__ void state_init_kernel(RState* st) { st->cur = 0; st->done = 0; st->cnt = 0.f; }

__global__ void commit_kernel(float* __restrict__ x, const float* __restrict__ y,
                              const RState* __restrict__ st)
{
  if (st->done) return;  // uses pre-update done (router runs after)
  int id = blockIdx.x * 256 + threadIdx.x;
  if (id >= T_SEQ * DMODEL) return;
  x[id] = y[id];
}

__global__ __launch_bounds__(256)
void router_kernel(const float* __restrict__ y, const float* __restrict__ Wr,
                   const float* __restrict__ br, const float* __restrict__ gu,
                   const int* __restrict__ mbcPtr, int stepi, RState* st)
{
  __shared__ float feat[RFEAT];
  __shared__ float logit[NROUTE];
  int cur = st->cur;
  for (int d = threadIdx.x; d < DMODEL; d += 256) {
    float sum = 0.f;
    for (int t = 0; t < T_SEQ; ++t) sum += y[(size_t)t * DMODEL + d];
    feat[d] = sum * (1.f / (float)T_SEQ);
  }
  if (threadIdx.x == 0) { feat[DMODEL] = st->cnt; feat[DMODEL + 1] = (float)mbcPtr[0]; }
  __syncthreads();
  if (threadIdx.x < NROUTE) {
    const float* w = Wr + ((size_t)cur * NROUTE + threadIdx.x) * RFEAT;
    float acc = br[cur * NROUTE + threadIdx.x];
    for (int d = 0; d < RFEAT; ++d) acc += feat[d] * w[d];
    float u = gu[stepi * NROUTE + threadIdx.x];
    float g = -__logf(-__logf(u + 1e-10f) + 1e-10f);
    logit[threadIdx.x] = acc + g;   // softmax/TAU monotone -> argmax of logits+g
  }
  __syncthreads();
  if (threadIdx.x == 0) {
    int nxt = 0; float best = logit[0];
    for (int j = 1; j < NROUTE; ++j) if (logit[j] > best) { best = logit[j]; nxt = j; }
    float mbc = (float)mbcPtr[0];
    int done = st->done;
    float ncnt = st->cnt + 1.f;
    int exitn = (nxt == NBLK) || (ncnt >= mbc);
    if (!done) st->cnt = ncnt;
    int ndone = done | exitn;
    if (!ndone) st->cur = nxt;
    st->done = ndone;
  }
}

// ---------------- host orchestration ----------------
extern "C" void kernel_launch(void* const* d_in, const int* in_sizes, int n_in,
                              void* d_out, int out_size, void* d_ws, size_t ws_size,
                              hipStream_t stream)
{
  (void)in_sizes; (void)n_in; (void)out_size; (void)ws_size;
  const int*   tokens    = (const int*)  d_in[0];
  const float* embed_w   = (const float*)d_in[1];
  const float* ve_w      = (const float*)d_in[2];
  const float* lm_head_w = (const float*)d_in[3];
  const float* in_wq     = (const float*)d_in[4];
  const float* in_wk     = (const float*)d_in[5];
  const float* in_wv     = (const float*)d_in[6];
  const float* in_wo     = (const float*)d_in[7];
  const float* in_alam   = (const float*)d_in[8];
  const float* in_fc     = (const float*)d_in[9];
  const float* in_pr     = (const float*)d_in[10];
  const float* in_lam    = (const float*)d_in[11];
  const float* mid_wq    = (const float*)d_in[12];
  const float* mid_wk    = (const float*)d_in[13];
  const float* mid_wv    = (const float*)d_in[14];
  const float* mid_wo    = (const float*)d_in[15];
  const float* mid_alam  = (const float*)d_in[16];
  const float* mid_fc    = (const float*)d_in[17];
  const float* mid_pr    = (const float*)d_in[18];
  const float* router_w  = (const float*)d_in[19];
  const float* router_b  = (const float*)d_in[20];
  const float* out_wq    = (const float*)d_in[21];
  const float* out_wk    = (const float*)d_in[22];
  const float* out_wv    = (const float*)d_in[23];
  const float* out_wo    = (const float*)d_in[24];
  const float* out_alam  = (const float*)d_in[25];
  const float* out_fc    = (const float*)d_in[26];
  const float* out_pr    = (const float*)d_in[27];
  const float* out_lam   = (const float*)d_in[28];
  const float* gumbel_u  = (const float*)d_in[29];
  const int*   wb_ptr    = (const int*)  d_in[30];
  const int*   mbc_ptr   = (const int*)  d_in[31];

  // ---- carve workspace ----
  char* wsp = (char*)d_ws;
  auto carve = [&](size_t bytes) {
    char* p = wsp; wsp += (bytes + 255) & ~(size_t)255; return p;
  };
  const size_t TD  = (size_t)T_SEQ * DMODEL;
  float*  cosb  = (float*) carve(T_SEQ * (HDIM / 2) * 4);
  float*  sinb  = (float*) carve(T_SEQ * (HDIM / 2) * 4);
  float*  x     = (float*) carve(TD * 4);
  float*  x0    = (float*) carve(TD * 4);
  float*  ve    = (float*) carve(TD * 4);
  float*  ybuf  = (float*) carve(TD * 4);
  float*  qf32  = (float*) carve(TD * 4);
  float*  kf32  = (float*) carve(TD * 4);
  float*  vf32  = (float*) carve(TD * 4);
  bf16_t* xn    = (bf16_t*)carve(TD * 2);
  bf16_t* qh    = (bf16_t*)carve(TD * 2);
  bf16_t* kh    = (bf16_t*)carve(TD * 2);
  bf16_t* vh    = (bf16_t*)carve(TD * 2);
  bf16_t* attnb = (bf16_t*)carve(TD * 2);
  bf16_t* act   = (bf16_t*)carve((size_t)T_SEQ * DFF * 2);
  RState* st    = (RState*)carve(sizeof(RState));
  const int* curP = (const int*)st;   // cur is first field

  const int EW_BLK = (int)((TD + 255) / 256);
  const dim3 gD(DMODEL / 64, T_SEQ / 256);     // GEMM -> D
  const dim3 gF(DFF / 64,    T_SEQ / 256);     // GEMM -> 4D
  const dim3 gV(NVOCAB / 64, T_SEQ / 256);     // GEMM -> vocab
  const dim3 gA(T_SEQ / 16, HHEADS);           // attention

  auto attn_block = [&](float* R, const float* veP,
                        const float* wq, const float* wk, const float* wv,
                        const float* wo, long long wstride, const int* cp,
                        const float* lamB) {
    rms_row_kernel<<<T_SEQ, 256, 0, stream>>>(R, xn, DMODEL);
    gemm_wmma_kernel<<<gD, 256, 0, stream>>>(xn, wq, wstride, cp, DMODEL, DMODEL, nullptr, qf32, 0);
    gemm_wmma_kernel<<<gD, 256, 0, stream>>>(xn, wk, wstride, cp, DMODEL, DMODEL, nullptr, kf32, 0);
    gemm_wmma_kernel<<<gD, 256, 0, stream>>>(xn, wv, wstride, cp, DMODEL, DMODEL, nullptr, vf32, 0);
    rope_head_kernel<<<T_SEQ * HHEADS, 32, 0, stream>>>(qf32, cosb, sinb, qh);
    rope_head_kernel<<<T_SEQ * HHEADS, 32, 0, stream>>>(kf32, cosb, sinb, kh);
    vprep_kernel<<<EW_BLK, 256, 0, stream>>>(vf32, veP, lamB, cp, vh);
    attn_kernel<<<gA, 32, 0, stream>>>(qh, kh, vh, attnb, wb_ptr);
    gemm_wmma_kernel<<<gD, 256, 0, stream>>>(attnb, wo, wstride, cp, DMODEL, DMODEL, R, R, 1);
  };
  auto mlp_block = [&](float* R, const float* fc, const float* pr,
                       long long fstride, const int* cp) {
    rms_row_kernel<<<T_SEQ, 256, 0, stream>>>(R, xn, DMODEL);
    gemm_wmma_kernel<<<gF, 256, 0, stream>>>(xn, fc, fstride, cp, DFF, DMODEL, nullptr, act, 2);
    gemm_wmma_kernel<<<gD, 256, 0, stream>>>(act, pr, fstride, cp, DMODEL, DFF, R, R, 1);
  };

  // ---- prologue ----
  rope_table_kernel<<<(T_SEQ * (HDIM / 2) + 63) / 64, 64, 0, stream>>>(cosb, sinb);
  embed_kernel<<<T_SEQ, 256, 0, stream>>>(tokens, embed_w, ve_w, x, x0, ve);

  // ---- input block ----
  mix_kernel<<<EW_BLK, 256, 0, stream>>>(x, x0, in_lam);
  attn_block(x, ve, in_wq, in_wk, in_wv, in_wo, 0, nullptr, in_alam);
  mlp_block(x, in_fc, in_pr, 0, nullptr);

  // ---- routed mid blocks (device-side state, 10 unrolled steps) ----
  state_init_kernel<<<1, 1, 0, stream>>>(st);
  const long long WS_ATT = (long long)DMODEL * DMODEL;  // stride per mid block
  const long long WS_MLP = (long long)DFF * DMODEL;
  for (int i = 0; i < NBLK; ++i) {
    hipMemcpyAsync(ybuf, x, TD * 4, hipMemcpyDeviceToDevice, stream);
    attn_block(ybuf, nullptr, mid_wq, mid_wk, mid_wv, mid_wo, WS_ATT, curP, mid_alam);
    mlp_block(ybuf, mid_fc, mid_pr, WS_MLP, curP);
    commit_kernel<<<EW_BLK, 256, 0, stream>>>(x, ybuf, st);   // uses old done
    router_kernel<<<1, 256, 0, stream>>>(ybuf, router_w, router_b, gumbel_u, mbc_ptr, i, st);
  }

  // ---- output block ----
  mix_kernel<<<EW_BLK, 256, 0, stream>>>(x, x0, out_lam);
  attn_block(x, ve, out_wq, out_wk, out_wv, out_wo, 0, nullptr, out_alam);
  mlp_block(x, out_fc, out_pr, 0, nullptr);

  // ---- lm head: rms(y) @ lm_head^T, 30*tanh(./30) ----
  rms_row_kernel<<<T_SEQ, 256, 0, stream>>>(x, xn, DMODEL);
  gemm_wmma_kernel<<<gV, 256, 0, stream>>>(xn, lm_head_w, 0, nullptr, NVOCAB, DMODEL,
                                           nullptr, d_out, 3);
}
